// LlamaAttention_82394652606926
// MI455X (gfx1250) — matrix operand order repaired and checked by reading, not compiled, not run on hardware
//
#include <hip/hip_runtime.h>
#include <hip/hip_bf16.h>
#include <stddef.h>

// ---------------------------------------------------------------------------
// MI455X (gfx1250) Llama attention.
//  - all GEMMs via v_wmma_f32_16x16x32_f16 (f32 accumulate)
//  - GEMM *and* attention K/V tiles staged into LDS by the Tensor Data Mover
//    (tensor_load_to_lds), double buffered, synced with s_wait_tensorcnt
// B=1, S=2048, H=2048, NH=32, NKV=8, HD=64, N_REP=4, scale=1/8
// ---------------------------------------------------------------------------

#define S_LEN 2048
#define HID   2048
#define NH    32
#define NKV   8
#define HD    64
#define QKV_COLS 3072   // 2048 Q + 512 K + 512 V

typedef __attribute__((ext_vector_type(16))) _Float16 v16h;
typedef __attribute__((ext_vector_type(8)))  _Float16 v8h;
typedef __attribute__((ext_vector_type(8)))  float    v8f;
typedef __attribute__((ext_vector_type(4)))  unsigned int u32x4;
typedef __attribute__((ext_vector_type(8)))  unsigned int u32x8;

union V16 { v16h v; v8h h8[2]; };

__device__ __forceinline__ v8f wmma_f16(v16h a, v16h b, v8f c) {
  // (neg_a, A, neg_b, B, c_mod, C, reuse_a, reuse_b)
  return __builtin_amdgcn_wmma_f32_16x16x32_f16(false, a, false, b, (short)0, c,
                                                false, false);
}

// ---------------------------------------------------------------------------
// TDM: DMA a 2D f16 tile (tile_d0 elems wide, tile_d1 rows, row stride
// stride_el elems) from global memory into LDS at byte offset lds_off.
// LDS pad: +16B after every 64B stored, so a 32-elem (64B) row lands with a
// 40-halfword stride and a 64-elem (128B) row with an 80-halfword stride
// where element e sits at (e>>5)*40 + (e&31).
// D# layout per CDNA5 ISA.
// ---------------------------------------------------------------------------
__device__ __forceinline__ void tdm_load_tile_f16(
    unsigned lds_off, const _Float16* gptr,
    unsigned tile_d0, unsigned tile_d1, unsigned long long stride_el) {
  const unsigned long long ga = (unsigned long long)(size_t)gptr;
  u32x4 g0;
  g0[0] = 1u;                                       // count=1 (valid user D#)
  g0[1] = lds_off;                                  // lds_addr
  g0[2] = (unsigned)(ga & 0xffffffffu);             // global_addr[31:0]
  g0[3] = (unsigned)((ga >> 32) & 0x01ffffffu)      // global_addr[56:32]
          | (2u << 30);                             // type=2 ("image")
  u32x8 g1;
  g1[0] = (1u << 16)                                // data_size=1 (2 bytes)
          | (1u << 20)                              // pad_enable
          | (3u << 22)                              // pad_interval: 16 DW (64B)
          | (3u << 25);                             // pad_amount: 4 DW (16B)
  g1[1] = (tile_d0 & 0xffffu) << 16;                // tensor_dim0[15:0]
  g1[2] = (tile_d0 >> 16) | ((tile_d1 & 0xffffu) << 16);   // dim0 hi | dim1 lo
  g1[3] = (tile_d1 >> 16) | ((tile_d0 & 0xffffu) << 16);   // dim1 hi | tile_dim0
  g1[4] = (tile_d1 & 0xffffu);                      // tile_dim1 (tile_dim2 = 0)
  g1[5] = (unsigned)(stride_el & 0xffffffffu);      // tensor_dim0_stride lo
  g1[6] = (unsigned)((stride_el >> 32) & 0xffffu);  // stride hi (dim1_stride=0)
  g1[7] = 0u;
  asm volatile("tensor_load_to_lds %0, %1" :: "s"(g0), "s"(g1) : "memory");
}

// ---------------------------------------------------------------------------
// f32 -> f16 conversion (grid-stride)
// ---------------------------------------------------------------------------
__global__ void cvt_f32_f16(const float* __restrict__ src,
                            _Float16* __restrict__ dst, int n) {
  for (int i = blockIdx.x * blockDim.x + threadIdx.x; i < n;
       i += gridDim.x * blockDim.x)
    dst[i] = (_Float16)src[i];
}

// ---------------------------------------------------------------------------
// C[M,N] = A[M,K] * B[N,K]^T   (f16, K-contiguous) — TDM-staged WMMA GEMM
// block: 256 thr = 8 waves (2 M x 4 N), block tile 64x128, wave tile 32x32
// double-buffered LDS tiles filled by the Tensor Data Mover (wave 0 issues).
// ---------------------------------------------------------------------------
template <typename OutT>
__global__ __launch_bounds__(256) void gemm_nt_kernel(
    const _Float16* __restrict__ A, const _Float16* __restrict__ B,
    OutT* __restrict__ C, int M, int N, int K) {
  __shared__ alignas(16) _Float16 lA[2][64][40];   // 40 = 32 + TDM pad
  __shared__ alignas(16) _Float16 lB[2][128][40];

  const int tid  = threadIdx.x;
  const int lane = tid & 31, wave = tid >> 5;
  const int half = lane >> 4, l16 = lane & 15;
  const int wm = wave >> 2, wn = wave & 3;

  const int bm = blockIdx.y * 64;
  const int bn = blockIdx.x * 128;

  const unsigned ldsA[2] = {(unsigned)(size_t)&lA[0][0][0],
                            (unsigned)(size_t)&lA[1][0][0]};
  const unsigned ldsB[2] = {(unsigned)(size_t)&lB[0][0][0],
                            (unsigned)(size_t)&lB[1][0][0]};

  v8f acc[2][2] = {};
  const int nk = K / 32;

  if (wave == 0) {   // prologue: fill buffer 0
    tdm_load_tile_f16(ldsA[0], &A[(size_t)bm * K], 32, 64, (unsigned)K);
    tdm_load_tile_f16(ldsB[0], &B[(size_t)bn * K], 32, 128, (unsigned)K);
  }

  for (int kt = 0; kt < nk; ++kt) {
    const int cur = kt & 1;
    if (kt + 1 < nk) {
      if (wave == 0) {   // kick next tile pair, then wait for current pair
        const int k1 = (kt + 1) * 32;
        tdm_load_tile_f16(ldsA[1 - cur], &A[(size_t)bm * K + k1], 32, 64,
                          (unsigned)K);
        tdm_load_tile_f16(ldsB[1 - cur], &B[(size_t)bn * K + k1], 32, 128,
                          (unsigned)K);
        __builtin_amdgcn_s_wait_tensorcnt(2);
      }
    } else if (wave == 0) {
      __builtin_amdgcn_s_wait_tensorcnt(0);
    }
    __syncthreads();   // buffer `cur` visible to all waves

    v16h af[2], bf[2];
#pragma unroll
    for (int mi = 0; mi < 2; ++mi) {   // A frag: lane = M row, k = half*8(+16)+i
      const int r = wm * 32 + mi * 16 + l16;
      V16 u;
      u.h8[0] = *(const v8h*)&lA[cur][r][half * 8];
      u.h8[1] = *(const v8h*)&lA[cur][r][16 + half * 8];
      af[mi] = u.v;
    }
#pragma unroll
    for (int ni = 0; ni < 2; ++ni) {   // B frag: lane = N col, k = half*8(+16)+i
      const int r = wn * 32 + ni * 16 + l16;
      V16 u;
      u.h8[0] = *(const v8h*)&lB[cur][r][half * 8];
      u.h8[1] = *(const v8h*)&lB[cur][r][16 + half * 8];
      bf[ni] = u.v;
    }
#pragma unroll
    for (int mi = 0; mi < 2; ++mi)
#pragma unroll
      for (int ni = 0; ni < 2; ++ni)
        acc[mi][ni] = wmma_f16(af[mi], bf[ni], acc[mi][ni]);
    __syncthreads();   // all done reading `cur`; TDM may overwrite it next iter
  }

#pragma unroll
  for (int mi = 0; mi < 2; ++mi)
#pragma unroll
    for (int ni = 0; ni < 2; ++ni)
#pragma unroll
      for (int r = 0; r < 8; ++r) {    // D layout: m = r + half*8, n = l16
        const int row = bm + wm * 32 + mi * 16 + r + half * 8;
        const int col = bn + wn * 32 + ni * 16 + l16;
        C[(size_t)row * N + col] = (OutT)acc[mi][ni][r];
      }
}

// ---------------------------------------------------------------------------
// RoPE + scatter into attention layouts.
//  QKVh[s][0..3071] (f16)  ->  Qh[h][s][d], Kh[kvh][s][d] (roped),
//                              VtT[kvh][d][s] (transposed, no rope)
// ---------------------------------------------------------------------------
__global__ void rope_scatter_kernel(const _Float16* __restrict__ qkv,
                                    const float* __restrict__ cosT,
                                    const float* __restrict__ sinT,
                                    _Float16* __restrict__ Qh,
                                    _Float16* __restrict__ Kh,
                                    _Float16* __restrict__ VtT) {
  const int i = blockIdx.x * blockDim.x + threadIdx.x;
  if (i >= S_LEN * QKV_COLS) return;
  const int col = i % QKV_COLS;
  const int s   = i / QKV_COLS;
  const size_t rowb = (size_t)s * QKV_COLS;

  if (col < NH * HD) {                       // Q
    const int h = col >> 6, d = col & 63;
    const float x  = (float)qkv[rowb + col];
    const int   pd = (d < 32) ? d + 32 : d - 32;
    const float xp = (float)qkv[rowb + (h << 6) + pd];
    const float c = cosT[s * HD + d], sn = sinT[s * HD + d];
    const float r = x * c + ((d < 32) ? -xp : xp) * sn;
    Qh[((size_t)h * S_LEN + s) * HD + d] = (_Float16)r;
  } else if (col < NH * HD + NKV * HD) {     // K
    const int cc = col - NH * HD;
    const int kh = cc >> 6, d = cc & 63;
    const float x  = (float)qkv[rowb + col];
    const int   pd = (d < 32) ? d + 32 : d - 32;
    const float xp = (float)qkv[rowb + NH * HD + (kh << 6) + pd];
    const float c = cosT[s * HD + d], sn = sinT[s * HD + d];
    const float r = x * c + ((d < 32) ? -xp : xp) * sn;
    Kh[((size_t)kh * S_LEN + s) * HD + d] = (_Float16)r;
  } else {                                   // V (transpose only)
    const int cc = col - NH * HD - NKV * HD;
    const int vh = cc >> 6, d = cc & 63;
    VtT[((size_t)vh * HD + d) * S_LEN + s] = qkv[rowb + col];
  }
}

// ---------------------------------------------------------------------------
// Flash attention, causal. grid = (S/64, NH), block = 128 thr = 4 waves.
// K/V 64x64 tiles are DMA'd into LDS by the TDM, double-buffered across key
// blocks: global latency for block kb+1 is hidden behind block kb's 16 WMMAs
// + softmax. Padded LDS rows: element e of a 64-elem row sits at
// (e>>5)*40 + (e&31)  (80-halfword row stride).
// ---------------------------------------------------------------------------
__global__ __launch_bounds__(128) void attn_kernel(
    const _Float16* __restrict__ Qh, const _Float16* __restrict__ Kh,
    const _Float16* __restrict__ VtT, _Float16* __restrict__ Ob) {
  __shared__ alignas(16) _Float16 lK[2][64][80];   // [key][d] tiles
  __shared__ alignas(16) _Float16 lV[2][64][80];   // [d][key] tiles
  __shared__ alignas(16) _Float16 lP[4][16][72];   // per-wave P slab

  const int tid  = threadIdx.x;
  const int lane = tid & 31, w = tid >> 5;
  const int half = lane >> 4, l16 = lane & 15;
  const int qblk = blockIdx.x, h = blockIdx.y;
  const int kvh  = h >> 2;             // N_REP = 4
  const int q0   = qblk * 64;
  const int qrow_base = q0 + w * 16 + half * 8;   // + r gives this lane's rows

  const _Float16* Kbase = Kh  + (size_t)kvh * S_LEN * HD;   // [key][d]
  const _Float16* Vbase = VtT + (size_t)kvh * HD * S_LEN;   // [d][key]

  const unsigned ldsK[2] = {(unsigned)(size_t)&lK[0][0][0],
                            (unsigned)(size_t)&lK[1][0][0]};
  const unsigned ldsV[2] = {(unsigned)(size_t)&lV[0][0][0],
                            (unsigned)(size_t)&lV[1][0][0]};

  // Q fragments: A layout, lane = row l16, reused across all key blocks
  v16h aq[2];
  {
    const _Float16* qp = Qh + ((size_t)h * S_LEN + q0 + w * 16 + l16) * HD;
#pragma unroll
    for (int kc = 0; kc < 2; ++kc) {
      V16 u;
      u.h8[0] = *(const v8h*)(qp + kc * 32 + half * 8);
      u.h8[1] = *(const v8h*)(qp + kc * 32 + 16 + half * 8);
      aq[kc] = u.v;
    }
  }

  v8f oacc[4] = {};
  float mrun[8], lrun[8];
#pragma unroll
  for (int r = 0; r < 8; ++r) { mrun[r] = -1e30f; lrun[r] = 0.0f; }

  const int nkb = qblk + 1;
  if (w == 0) {   // prologue: stage key block 0
    tdm_load_tile_f16(ldsK[0], Kbase, 64, 64, HD);
    tdm_load_tile_f16(ldsV[0], Vbase, 64, 64, S_LEN);
  }

  for (int kb = 0; kb < nkb; ++kb) {
    const int cur  = kb & 1;
    const int key0 = kb * 64;

    if (kb + 1 < nkb) {
      if (w == 0) {   // kick key block kb+1, wait for kb's pair
        tdm_load_tile_f16(ldsK[1 - cur], Kbase + (size_t)(key0 + 64) * HD,
                          64, 64, HD);
        tdm_load_tile_f16(ldsV[1 - cur], Vbase + (key0 + 64),
                          64, 64, S_LEN);
        __builtin_amdgcn_s_wait_tensorcnt(2);
      }
    } else if (w == 0) {
      __builtin_amdgcn_s_wait_tensorcnt(0);
    }
    __syncthreads();   // K/V tiles for `kb` visible to all waves

    // ---- S = Q K^T : B frag (nt,kc): row=key, d-chunks at (kc*40 + half*8 (+16)) -
    float sv[4][8];
#pragma unroll
    for (int nt = 0; nt < 4; ++nt) {
      const _Float16* kp = &lK[cur][nt * 16 + l16][0];
      v8f z = {};
#pragma unroll
      for (int kc = 0; kc < 2; ++kc) {
        V16 u;
        u.h8[0] = *(const v8h*)(kp + kc * 40 + half * 8);
        u.h8[1] = *(const v8h*)(kp + kc * 40 + 16 + half * 8);
        z = wmma_f16(aq[kc], u.v, z);
      }
      const int key = key0 + nt * 16 + l16;
#pragma unroll
      for (int r = 0; r < 8; ++r) {
        const float x = z[r] * 0.125f;                  // HD^-0.5
        sv[nt][r] = (key > qrow_base + r) ? -1e30f : x; // causal
      }
    }

    // ---- online softmax (row stats replicated across each 16-lane group) --
    float mnew[8], alpha[8];
#pragma unroll
    for (int r = 0; r < 8; ++r) {
      float m = mrun[r];
#pragma unroll
      for (int nt = 0; nt < 4; ++nt) m = fmaxf(m, sv[nt][r]);
      for (int off = 8; off >= 1; off >>= 1)
        m = fmaxf(m, __shfl_xor(m, off, 32));
      mnew[r]  = m;
      alpha[r] = __expf(mrun[r] - m);
      mrun[r]  = m;
    }
#pragma unroll
    for (int r = 0; r < 8; ++r) {
      float s = 0.0f;
#pragma unroll
      for (int nt = 0; nt < 4; ++nt) {
        const float p = __expf(sv[nt][r] - mnew[r]);
        sv[nt][r] = p;
        s += p;
      }
      for (int off = 8; off >= 1; off >>= 1) s += __shfl_xor(s, off, 32);
      lrun[r] = lrun[r] * alpha[r] + s;
    }

    // scale running accumulator
#pragma unroll
    for (int nt = 0; nt < 4; ++nt)
#pragma unroll
      for (int r = 0; r < 8; ++r) oacc[nt][r] *= alpha[r];

    // ---- P: D-layout -> A-layout via per-wave LDS (DS ops are in-order) ---
#pragma unroll
    for (int nt = 0; nt < 4; ++nt)
#pragma unroll
      for (int r = 0; r < 8; ++r)
        lP[w][half * 8 + r][nt * 16 + l16] = (_Float16)sv[nt][r];

    v16h ap[2];
#pragma unroll
    for (int kc = 0; kc < 2; ++kc) {
      V16 u;
      u.h8[0] = *(const v8h*)&lP[w][l16][kc * 32 + half * 8];
      u.h8[1] = *(const v8h*)&lP[w][l16][kc * 32 + 16 + half * 8];
      ap[kc] = u.v;
    }

    // ---- O += P * V : B frag (ntd,kc): row=d, key-chunks at (kc*40+half*8 (+16)) -
#pragma unroll
    for (int ntd = 0; ntd < 4; ++ntd) {
      const _Float16* vp = &lV[cur][ntd * 16 + l16][0];
#pragma unroll
      for (int kc = 0; kc < 2; ++kc) {
        V16 u;
        u.h8[0] = *(const v8h*)(vp + kc * 40 + half * 8);
        u.h8[1] = *(const v8h*)(vp + kc * 40 + 16 + half * 8);
        oacc[ntd] = wmma_f16(ap[kc], u.v, oacc[ntd]);
      }
    }
    __syncthreads();   // everyone done with tile `cur`; TDM may overwrite it
  }

  // ---- epilogue: normalize and emit O[s][h*64+d] as f16 ----
#pragma unroll
  for (int r = 0; r < 8; ++r) {
    const float inv = 1.0f / lrun[r];
    const int row = qrow_base + r;
#pragma unroll
    for (int ntd = 0; ntd < 4; ++ntd)
      Ob[(size_t)row * (NH * HD) + h * HD + ntd * 16 + l16] =
          (_Float16)(oacc[ntd][r] * inv);
  }
}

// ---------------------------------------------------------------------------
// Host-side orchestration
// ---------------------------------------------------------------------------
extern "C" void kernel_launch(void* const* d_in, const int* in_sizes, int n_in,
                              void* d_out, int out_size, void* d_ws,
                              size_t ws_size, hipStream_t stream) {
  (void)in_sizes; (void)n_in; (void)out_size; (void)ws_size;

  const float* x    = (const float*)d_in[0];   // hidden_states (S,H)
  const float* cosT = (const float*)d_in[1];   // (1,S,HD)
  const float* sinT = (const float*)d_in[2];   // (1,S,HD)
  // d_in[3] = attention_mask (causal) — recomputed on device
  const float* wq = (const float*)d_in[4];     // (2048,2048)
  const float* wk = (const float*)d_in[5];     // (512,2048)
  const float* wv = (const float*)d_in[6];     // (512,2048)
  const float* wo = (const float*)d_in[7];     // (2048,2048)
  float* out = (float*)d_out;

  // workspace carve-up (f16 elements), all 16B-aligned
  _Float16* Xh   = (_Float16*)d_ws;                    // S*H
  _Float16* Wcat = Xh   + (size_t)S_LEN * HID;         // 3072*2048 (Wq|Wk|Wv)
  _Float16* Woh  = Wcat + (size_t)QKV_COLS * HID;      // 2048*2048
  _Float16* QKVh = Woh  + (size_t)HID * (NH * HD);     // S*3072
  _Float16* Qh   = QKVh + (size_t)S_LEN * QKV_COLS;    // 32*2048*64
  _Float16* Kh   = Qh   + (size_t)NH * S_LEN * HD;     // 8*2048*64
  _Float16* VtT  = Kh   + (size_t)NKV * S_LEN * HD;    // 8*64*2048
  _Float16* Ob   = VtT  + (size_t)NKV * HD * S_LEN;    // S*2048

  const int CB = 256;
  auto cvt = [&](const float* s, _Float16* d, int n) {
    cvt_f32_f16<<<(n + CB - 1) / CB, CB, 0, stream>>>(s, d, n);
  };
  cvt(x,  Xh,                                   S_LEN * HID);
  cvt(wq, Wcat,                                 NH * HD * HID);
  cvt(wk, Wcat + (size_t)NH * HD * HID,         NKV * HD * HID);
  cvt(wv, Wcat + (size_t)(NH + NKV) * HD * HID, NKV * HD * HID);
  cvt(wo, Woh,                                  HID * NH * HD);

  // QKV = X * Wcat^T   (2048x2048)x(3072x2048)^T
  gemm_nt_kernel<_Float16>
      <<<dim3(QKV_COLS / 128, S_LEN / 64), 256, 0, stream>>>(
          Xh, Wcat, QKVh, S_LEN, QKV_COLS, HID);

  // RoPE + scatter into attention layouts
  {
    const int n = S_LEN * QKV_COLS;
    rope_scatter_kernel<<<(n + CB - 1) / CB, CB, 0, stream>>>(
        QKVh, cosT, sinT, Qh, Kh, VtT);
  }

  // causal flash attention
  attn_kernel<<<dim3(S_LEN / 64, NH), 128, 0, stream>>>(Qh, Kh, VtT, Ob);

  // out = O * Wo^T -> f32
  gemm_nt_kernel<float>
      <<<dim3(HID / 128, S_LEN / 64), 256, 0, stream>>>(
          Ob, Woh, out, S_LEN, HID, NH * HD);
}
